// ValueNetwork_83210696392993
// MI455X (gfx1250) — compile-verified, matrix operand order checked by reading, not run on hardware
//
#include <hip/hip_runtime.h>
#include <hip/hip_bf16.h>
#include <stdint.h>

#define N_NODES 50000
#define N_EDGES 800000
#define F_NODE 64
#define F_EDGE 32
#define H1 256
#define H2 128

typedef __attribute__((ext_vector_type(16))) __bf16 v16bf;
typedef __attribute__((ext_vector_type(8)))  float  v8f;
typedef __attribute__((ext_vector_type(16))) unsigned short v16u;

union ABVec {
    v16u u;
    v16bf b;
    uint4 q[2];
    unsigned short s[16];
};

__device__ __forceinline__ unsigned short f2bf(float f) {
    unsigned int x = __float_as_uint(f);
    x += 0x7FFFu + ((x >> 16) & 1u);   // round-to-nearest-even
    return (unsigned short)(x >> 16);
}

// ---- CDNA5 async global->LDS copy (ASYNCcnt-tracked), 16B per lane ----
__device__ __forceinline__ void async_copy16(unsigned lds_off, const void* gaddr) {
    asm volatile("global_load_async_to_lds_b128 %0, %1, off"
                 :: "v"(lds_off), "v"(gaddr) : "memory");
}
__device__ __forceinline__ void wait_async0() {
    asm volatile("s_wait_asynccnt 0" ::: "memory");
}

// Stage the nt'th B panel (KT K-tiles = KT*1024 bytes) into LDS at dst_off.
template <int KT>
__device__ __forceinline__ void stage_panel(const unsigned short* __restrict__ P,
                                            int nt, unsigned dst_off, int tid, int nth) {
    const char* src = (const char*)P + (size_t)nt * (KT * 1024);
    for (int c = tid; c < KT * 64; c += nth)
        async_copy16(dst_off + c * 16, src + (size_t)c * 16);
}

// A-operand (16x32 bf16, M x K): lane L holds row M = L%16.
// half = L/16 selects K runs [half*8, half*8+7] and [16+half*8, 16+half*8+7].
__device__ __forceinline__ v16bf load_a_bf(const unsigned short* __restrict__ row,
                                           int kbase, int half) {
    ABVec v;
    v.q[0] = *(const uint4*)(row + kbase + half * 8);
    v.q[1] = *(const uint4*)(row + kbase + 16 + half * 8);
    return v.b;
}

__device__ __forceinline__ v16bf load_a_f32(const float* __restrict__ row,
                                            int kbase, int half, float scale) {
    ABVec v;
    const float* r0 = row + kbase + half * 8;
    const float* r1 = row + kbase + 16 + half * 8;
#pragma unroll
    for (int i = 0; i < 8; i++) {
        v.s[i]     = f2bf(r0[i] * scale);
        v.s[8 + i] = f2bf(r1[i] * scale);
    }
    return v.b;
}

__device__ __forceinline__ v8f wmma_bf(v16bf a, v16bf b, v8f c) {
    return __builtin_amdgcn_wmma_f32_16x16x32_bf16(false, a, false, b,
                                                   (short)0, c, false, false);
}

// ---------------- prep kernels ----------------

__global__ void k_degree(const int* __restrict__ recv, float* __restrict__ deg, int n) {
    int i = blockIdx.x * blockDim.x + threadIdx.x;
    if (i < n) atomicAdd(&deg[recv[i]], 1.0f);
}

__global__ void k_invdeg(const float* __restrict__ deg, float* __restrict__ inv, int n) {
    int i = blockIdx.x * blockDim.x + threadIdx.x;
    if (i < n) {
        float d = deg[i];
        inv[i] = d > 0.f ? 1.f / fmaxf(d, 1.f) : 0.f;
    }
}

__global__ void k_f2bf(const float* __restrict__ in, unsigned short* __restrict__ out, int n) {
    int i = blockIdx.x * blockDim.x + threadIdx.x;
    if (i < n) out[i] = f2bf(in[i]);
}

// Pack fp32 W[Ksrc x N] (row-major) into WMMA B-operand tile order at K-tile
// offset kt_base within a stacked packed buffer with KT_total K-tiles.
__global__ void k_pack(const float* __restrict__ W, unsigned short* __restrict__ dst,
                       int N, int Ksrc, int KT_total, int kt_base) {
    int tid = blockIdx.x * blockDim.x + threadIdx.x;
    int KTs = Ksrc / 32;
    int total = (N / 16) * KTs * 32;
    if (tid >= total) return;
    int lane = tid & 31;
    int t    = tid >> 5;
    int ktl  = t % KTs;
    int nt   = t / KTs;
    int half = lane >> 4;
    int n    = nt * 16 + (lane & 15);
    unsigned short* out =
        dst + (((size_t)(nt * KT_total + kt_base + ktl) * 32 + lane) << 4);
#pragma unroll
    for (int e = 0; e < 16; e++) {
        int k = ktl * 32 + ((e < 8) ? (half * 8 + e) : (16 + half * 8 + (e - 8)));
        out[e] = f2bf(W[(size_t)k * N + n]);
    }
}

// ---------------- layer 1: edges ----------------
// e1 = relu([edges | nodes[s] | nodes[r]] @ [We1;Ws1;Wr1] + be1)   K=160, N=256
// B double-buffered in LDS via async copies; fused segment-sum into acc1.
__global__ void __launch_bounds__(128)
k_edge1(const float* __restrict__ edges, const unsigned short* __restrict__ nodesb,
        const int* __restrict__ senders, const int* __restrict__ receivers,
        const unsigned short* __restrict__ P1, const float* __restrict__ be1,
        unsigned short* __restrict__ e1, float* __restrict__ acc1) {
    __shared__ __align__(16) char smem[2 * 5 * 1024];
    const int tid  = threadIdx.x;
    const int gw   = (blockIdx.x * blockDim.x + tid) >> 5;   // grid exact: 50000 waves
    const int lane = tid & 31;
    const int base = gw * 16;
    const int half = lane >> 4;
    const int mrow = lane & 15;
    const unsigned lbase = (unsigned)(uintptr_t)&smem[0];

    stage_panel<5>(P1, 0, lbase, tid, 128);

    v16bf A[5];
    A[0] = load_a_f32(edges + (size_t)(base + mrow) * F_EDGE, 0, half, 1.f);
    const unsigned short* sn = nodesb + (size_t)senders[base + mrow] * F_NODE;
    const unsigned short* rn = nodesb + (size_t)receivers[base + mrow] * F_NODE;
    A[1] = load_a_bf(sn, 0, half);
    A[2] = load_a_bf(sn, 32, half);
    A[3] = load_a_bf(rn, 0, half);
    A[4] = load_a_bf(rn, 32, half);

    int rj[8];
#pragma unroll
    for (int j = 0; j < 8; j++) rj[j] = receivers[base + half * 8 + j];

    int cur = 0;
    for (int nt = 0; nt < 16; nt++) {
        wait_async0();
        __syncthreads();
        if (nt + 1 < 16) stage_panel<5>(P1, nt + 1, lbase + (cur ^ 1) * 5120, tid, 128);

        const ABVec* bp = (const ABVec*)(smem + cur * 5120);
        v8f acc = {};
#pragma unroll
        for (int kt = 0; kt < 5; kt++)
            acc = wmma_bf(A[kt], bp[(kt << 5) + lane].b, acc);

        int col = nt * 16 + mrow;
        float bias = be1[col];
#pragma unroll
        for (int j = 0; j < 8; j++) {
            float v = fmaxf(acc[j] + bias, 0.f);
            int m = half * 8 + j;
            e1[(size_t)(base + m) * H1 + col] = f2bf(v);
            atomicAdd(&acc1[(size_t)rj[j] * H1 + col], v);
        }
        __syncthreads();
        cur ^= 1;
    }
}

// ---------------- layer 1: nodes ----------------
// n1 = relu([nodes | acc1*invdeg] @ [Wn1;Win1] + bn1)   K=320, N=256
__global__ void __launch_bounds__(160)
k_node1(const unsigned short* __restrict__ nodesb, const float* __restrict__ acc1,
        const float* __restrict__ invdeg, const unsigned short* __restrict__ PN1,
        const float* __restrict__ bn1, unsigned short* __restrict__ n1) {
    __shared__ __align__(16) char smem[2 * 10 * 1024];
    const int tid  = threadIdx.x;
    const int gw   = (blockIdx.x * blockDim.x + tid) >> 5;   // grid exact: 3125 waves
    const int lane = tid & 31;
    const int base = gw * 16;
    const int half = lane >> 4;
    const int mrow = lane & 15;
    const int row  = base + mrow;
    const unsigned lbase = (unsigned)(uintptr_t)&smem[0];

    stage_panel<10>(PN1, 0, lbase, tid, 160);

    float sc = invdeg[row];
    v16bf A[10];
    A[0] = load_a_bf(nodesb + (size_t)row * F_NODE, 0, half);
    A[1] = load_a_bf(nodesb + (size_t)row * F_NODE, 32, half);
    const float* ar = acc1 + (size_t)row * H1;
#pragma unroll
    for (int kt = 0; kt < 8; kt++) A[2 + kt] = load_a_f32(ar, kt * 32, half, sc);

    int cur = 0;
    for (int nt = 0; nt < 16; nt++) {
        wait_async0();
        __syncthreads();
        if (nt + 1 < 16) stage_panel<10>(PN1, nt + 1, lbase + (cur ^ 1) * 10240, tid, 160);

        const ABVec* bp = (const ABVec*)(smem + cur * 10240);
        v8f acc = {};
#pragma unroll
        for (int kt = 0; kt < 10; kt++)
            acc = wmma_bf(A[kt], bp[(kt << 5) + lane].b, acc);

        int col = nt * 16 + mrow;
        float bias = bn1[col];
#pragma unroll
        for (int j = 0; j < 8; j++) {
            float v = fmaxf(acc[j] + bias, 0.f);
            n1[(size_t)(base + half * 8 + j) * H1 + col] = f2bf(v);
        }
        __syncthreads();
        cur ^= 1;
    }
}

// ---------------- layer 2: edges ----------------
// e2 = relu([e1 | n1[s] | n1[r]] @ [We2;Ws2;Wr2] + be2)   K=768, N=128
// e2 never materialized: fused atomic segment-sum (acc2) + column-sum (e2sum).
__global__ void __launch_bounds__(128)
k_edge2(const unsigned short* __restrict__ e1, const unsigned short* __restrict__ n1,
        const int* __restrict__ senders, const int* __restrict__ receivers,
        const unsigned short* __restrict__ P2, const float* __restrict__ be2,
        float* __restrict__ acc2, float* __restrict__ e2sum) {
    __shared__ __align__(16) char smem[2 * 24 * 1024];
    __shared__ float ls[H2];
    const int tid = threadIdx.x;
    ls[tid & 127] = 0.f;

    const int gw   = (blockIdx.x * blockDim.x + tid) >> 5;   // grid exact: 50000 waves
    const int lane = tid & 31;
    const int base = gw * 16;
    const int half = lane >> 4;
    const int mrow = lane & 15;
    const unsigned lbase = (unsigned)(uintptr_t)&smem[0];

    stage_panel<24>(P2, 0, lbase, tid, 128);

    v16bf A[24];
    const unsigned short* er = e1 + (size_t)(base + mrow) * H1;
    const unsigned short* sn = n1 + (size_t)senders[base + mrow] * H1;
    const unsigned short* rn = n1 + (size_t)receivers[base + mrow] * H1;
#pragma unroll
    for (int kt = 0; kt < 8; kt++) {
        A[kt]      = load_a_bf(er, kt * 32, half);
        A[8 + kt]  = load_a_bf(sn, kt * 32, half);
        A[16 + kt] = load_a_bf(rn, kt * 32, half);
    }
    int rj[8];
#pragma unroll
    for (int j = 0; j < 8; j++) rj[j] = receivers[base + half * 8 + j];

    int cur = 0;
    for (int nt = 0; nt < 8; nt++) {
        wait_async0();
        __syncthreads();
        if (nt + 1 < 8) stage_panel<24>(P2, nt + 1, lbase + (cur ^ 1) * 24576, tid, 128);

        const ABVec* bp = (const ABVec*)(smem + cur * 24576);
        v8f acc = {};
#pragma unroll
        for (int kt = 0; kt < 24; kt++)
            acc = wmma_bf(A[kt], bp[(kt << 5) + lane].b, acc);

        int col = nt * 16 + mrow;
        float bias = be2[col];
        float csum = 0.f;
#pragma unroll
        for (int j = 0; j < 8; j++) {
            float v = fmaxf(acc[j] + bias, 0.f);
            atomicAdd(&acc2[(size_t)rj[j] * H2 + col], v);
            csum += v;
        }
        atomicAdd(&ls[col], csum);
        __syncthreads();
        cur ^= 1;
    }
    if (tid < H2) atomicAdd(&e2sum[tid], ls[tid]);
}

// ---------------- layer 2: nodes ----------------
// n2 = relu([n1 | acc2*invdeg] @ [Wn2;Win2] + bn2)   K=384, N=128
// n2 never materialized: only its column sums survive (for the mean).
__global__ void __launch_bounds__(160)
k_node2(const unsigned short* __restrict__ n1, const float* __restrict__ acc2,
        const float* __restrict__ invdeg, const unsigned short* __restrict__ PN2,
        const float* __restrict__ bn2, float* __restrict__ n2sum) {
    __shared__ __align__(16) char smem[2 * 12 * 1024];
    __shared__ float ls[H2];
    const int tid = threadIdx.x;
    if (tid < H2) ls[tid] = 0.f;

    const int gw   = (blockIdx.x * blockDim.x + tid) >> 5;   // grid exact: 3125 waves
    const int lane = tid & 31;
    const int base = gw * 16;
    const int half = lane >> 4;
    const int mrow = lane & 15;
    const int row  = base + mrow;
    const unsigned lbase = (unsigned)(uintptr_t)&smem[0];

    stage_panel<12>(PN2, 0, lbase, tid, 160);

    float sc = invdeg[row];
    v16bf A[12];
    const unsigned short* nr = n1 + (size_t)row * H1;
#pragma unroll
    for (int kt = 0; kt < 8; kt++) A[kt] = load_a_bf(nr, kt * 32, half);
    const float* ar = acc2 + (size_t)row * H2;
#pragma unroll
    for (int kt = 0; kt < 4; kt++) A[8 + kt] = load_a_f32(ar, kt * 32, half, sc);

    int cur = 0;
    for (int nt = 0; nt < 8; nt++) {
        wait_async0();
        __syncthreads();
        if (nt + 1 < 8) stage_panel<12>(PN2, nt + 1, lbase + (cur ^ 1) * 12288, tid, 160);

        const ABVec* bp = (const ABVec*)(smem + cur * 12288);
        v8f acc = {};
#pragma unroll
        for (int kt = 0; kt < 12; kt++)
            acc = wmma_bf(A[kt], bp[(kt << 5) + lane].b, acc);

        int col = nt * 16 + mrow;
        float bias = bn2[col];
        float csum = 0.f;
#pragma unroll
        for (int j = 0; j < 8; j++) csum += fmaxf(acc[j] + bias, 0.f);
        atomicAdd(&ls[col], csum);
        __syncthreads();
        cur ^= 1;
    }
    if (tid < H2) atomicAdd(&n2sum[tid], ls[tid]);
}

// ---------------- global readout ----------------
__global__ void k_final(const float* __restrict__ n2sum, const float* __restrict__ e2sum,
                        const float* __restrict__ Wgn, const float* __restrict__ Wge,
                        const float* __restrict__ bg, float* __restrict__ out) {
    __shared__ float red[H2];
    int t = threadIdx.x;
    float v = (n2sum[t] * (1.0f / N_NODES)) * Wgn[t] +
              (e2sum[t] * (1.0f / N_EDGES)) * Wge[t];
    red[t] = v;
    __syncthreads();
    for (int s = 64; s > 0; s >>= 1) {
        if (t < s) red[t] += red[t + s];
        __syncthreads();
    }
    if (t == 0) out[0] = red[0] + bg[0];
}

// ---------------- host launch ----------------

extern "C" void kernel_launch(void* const* d_in, const int* in_sizes, int n_in,
                              void* d_out, int out_size, void* d_ws, size_t ws_size,
                              hipStream_t stream) {
    const float* nodes     = (const float*)d_in[0];
    const float* edges     = (const float*)d_in[1];
    const int*   senders   = (const int*)d_in[2];
    const int*   receivers = (const int*)d_in[3];
    const float* We1 = (const float*)d_in[4];
    const float* Ws1 = (const float*)d_in[5];
    const float* Wr1 = (const float*)d_in[6];
    const float* be1 = (const float*)d_in[7];
    const float* Wn1 = (const float*)d_in[8];
    const float* Win1 = (const float*)d_in[9];
    const float* bn1 = (const float*)d_in[10];
    const float* We2 = (const float*)d_in[11];
    const float* Ws2 = (const float*)d_in[12];
    const float* Wr2 = (const float*)d_in[13];
    const float* be2 = (const float*)d_in[14];
    const float* Wn2 = (const float*)d_in[15];
    const float* Win2 = (const float*)d_in[16];
    const float* bn2 = (const float*)d_in[17];
    const float* Wgn = (const float*)d_in[18];
    const float* Wge = (const float*)d_in[19];
    const float* bg  = (const float*)d_in[20];

    char* ws = (char*)d_ws;
    size_t off = 0;
    auto carve = [&](size_t bytes) -> void* {
        void* p = ws + off;
        off += (bytes + 511) & ~(size_t)511;
        return p;
    };
    float*          deg    = (float*)carve((size_t)N_NODES * 4);
    float*          invdeg = (float*)carve((size_t)N_NODES * 4);
    unsigned short* nodesb = (unsigned short*)carve((size_t)N_NODES * F_NODE * 2);
    unsigned short* P1     = (unsigned short*)carve((size_t)16 * 5 * 32 * 16 * 2);
    unsigned short* PN1    = (unsigned short*)carve((size_t)16 * 10 * 32 * 16 * 2);
    unsigned short* P2     = (unsigned short*)carve((size_t)8 * 24 * 32 * 16 * 2);
    unsigned short* PN2    = (unsigned short*)carve((size_t)8 * 12 * 32 * 16 * 2);
    unsigned short* e1     = (unsigned short*)carve((size_t)N_EDGES * H1 * 2);
    unsigned short* n1     = (unsigned short*)carve((size_t)N_NODES * H1 * 2);
    float*          acc1   = (float*)carve((size_t)N_NODES * H1 * 4);
    float*          acc2   = (float*)carve((size_t)N_NODES * H2 * 4);
    float*          e2sum  = (float*)carve((size_t)H2 * 4);
    float*          n2sum  = (float*)carve((size_t)H2 * 4);

    hipMemsetAsync(deg, 0, (size_t)N_NODES * 4, stream);
    hipMemsetAsync(acc1, 0, (size_t)N_NODES * H1 * 4, stream);
    hipMemsetAsync(acc2, 0, (size_t)N_NODES * H2 * 4, stream);
    hipMemsetAsync(e2sum, 0, (size_t)H2 * 4, stream);
    hipMemsetAsync(n2sum, 0, (size_t)H2 * 4, stream);

    k_degree<<<(N_EDGES + 255) / 256, 256, 0, stream>>>(receivers, deg, N_EDGES);
    k_invdeg<<<(N_NODES + 255) / 256, 256, 0, stream>>>(deg, invdeg, N_NODES);
    k_f2bf<<<((N_NODES * F_NODE) + 255) / 256, 256, 0, stream>>>(nodes, nodesb,
                                                                 N_NODES * F_NODE);

    auto pack = [&](const float* W, unsigned short* dst, int N, int Ksrc,
                    int KT_total, int kt_base) {
        int total = (N / 16) * (Ksrc / 32) * 32;
        k_pack<<<(total + 255) / 256, 256, 0, stream>>>(W, dst, N, Ksrc, KT_total, kt_base);
    };
    pack(We1, P1, H1, 32, 5, 0);
    pack(Ws1, P1, H1, 64, 5, 1);
    pack(Wr1, P1, H1, 64, 5, 3);
    pack(Wn1, PN1, H1, 64, 10, 0);
    pack(Win1, PN1, H1, 256, 10, 2);
    pack(We2, P2, H2, 256, 24, 0);
    pack(Ws2, P2, H2, 256, 24, 8);
    pack(Wr2, P2, H2, 256, 24, 16);
    pack(Wn2, PN2, H2, 256, 12, 0);
    pack(Win2, PN2, H2, 128, 12, 8);

    k_edge1<<<N_EDGES / 16 / 4, 128, 0, stream>>>(edges, nodesb, senders, receivers,
                                                  P1, be1, e1, acc1);
    k_node1<<<N_NODES / 16 / 5, 160, 0, stream>>>(nodesb, acc1, invdeg, PN1, bn1, n1);
    k_edge2<<<N_EDGES / 16 / 4, 128, 0, stream>>>(e1, n1, senders, receivers,
                                                  P2, be2, acc2, e2sum);
    k_node2<<<N_NODES / 16 / 5, 160, 0, stream>>>(n1, acc2, invdeg, PN2, bn2, n2sum);
    k_final<<<1, 128, 0, stream>>>(n2sum, e2sum, Wgn, Wge, bg, (float*)d_out);
}
